// LSTM_89361089560808
// MI455X (gfx1250) — matrix operand order repaired and checked by reading, not compiled, not run on hardware
//
#include <hip/hip_runtime.h>
#include <stdint.h>

// CDNA5 / gfx1250 WMMA LSTM.
//
// Latency-bound sequential recurrence: 2 blocks x 1 wave32, each wave owns 16
// independent batch rows. Gates computed transposed (G^T = W_hh x h^T) so the
// WMMA C/D layout puts batch in lanes and gates in VGPR rows -> the whole
// nonlinear cell update is lane-local. h is kept as a bf16 hi+lo split (two
// chained WMMAs per tile) to preserve ~fp32 accuracy across 65536 steps.

typedef __attribute__((ext_vector_type(16))) __bf16 v16bf;
typedef __attribute__((ext_vector_type(8)))  float  v8f;

#define NBATCH 16  // batches per block (one wave)

// LDS layout (bytes), per block
#define LDS_A    0        // W_hh bf16 [128][32] row-major      : 8192 B
#define LDS_BIAS 8192     // (b_ih + b_hh) f32 [128]            : 512 B
#define LDS_WIH  8704     // W_ih f32 [128]                     : 512 B
#define LDS_HHI  9216     // h_hi bf16 [16 batch][32 hidden]    : 1024 B
#define LDS_HLO  10240    // h_lo bf16 [16 batch][32 hidden]    : 1024 B
#define LDS_SIZE 11264

union V16U { v16bf v; uint4 q[2]; };
union V8U  { v8f   v; uint4 q[2]; float f[8]; };
union HPK  { __bf16 b[8]; uint4 q; };

__device__ __forceinline__ float fsig(float x) {
  // 1 / (1 + 2^(-x*log2e))  -> v_exp_f32 + v_rcp_f32 (trans pipe)
  return __builtin_amdgcn_rcpf(1.0f + __builtin_amdgcn_exp2f(-1.44269504f * x));
}
__device__ __forceinline__ float ftanh(float x) {
  // 2*sigmoid(2x) - 1
  return fmaf(2.0f, __builtin_amdgcn_rcpf(1.0f + __builtin_amdgcn_exp2f(-2.88539008f * x)), -1.0f);
}

__global__ __launch_bounds__(32)
void lstm_wmma_kernel(const float* __restrict__ x,
                      const float* __restrict__ W_ih,
                      const float* __restrict__ W_hh,
                      const float* __restrict__ b_ih,
                      const float* __restrict__ b_hh,
                      const float* __restrict__ W_lin,
                      const float* __restrict__ b_lin,
                      float* __restrict__ out,
                      int S) {
  __shared__ __align__(16) unsigned char lds[LDS_SIZE];

  const int lane = (int)threadIdx.x;   // 0..31 (wave32)
  const int n    = lane & 15;          // local batch / N column
  const int half = lane >> 4;          // C-layout M-offset selector

  // ---------------- one-time setup (single wave; DS is in-order) -----------
  // W_hh (128x32 f32, row-major [gate][hidden]) -> bf16 plane, A-layout addressable
  for (int i = lane; i < 128 * 32; i += 32)
    ((__bf16*)(lds + LDS_A))[i] = (__bf16)W_hh[i];
  for (int i = lane; i < 128; i += 32) {
    ((float*)(lds + LDS_BIAS))[i] = b_ih[i] + b_hh[i];
    ((float*)(lds + LDS_WIH))[i]  = W_ih[i];          // C == 1
  }
  for (int i = lane; i < 2048 / 4; i += 32)            // zero h_hi + h_lo planes
    ((uint32_t*)(lds + LDS_HHI))[i] = 0u;

  // per-lane head weights: lane's 16 hidden units are j = 16*t + 8*half + r
  float wlinv[16];
#pragma unroll
  for (int t = 0; t < 2; ++t)
#pragma unroll
    for (int r = 0; r < 8; ++r)
      wlinv[t * 8 + r] = W_lin[16 * t + 8 * half + r];
  const float blin = b_lin[0];

  float cst[2][8];                                     // cell state (lane-local)
#pragma unroll
  for (int g = 0; g < 2; ++g)
#pragma unroll
    for (int r = 0; r < 8; ++r) cst[g][r] = 0.0f;

  const int    batch = (int)blockIdx.x * NBATCH + n;
  const float* xrow  = x   + (size_t)batch * (size_t)S;
  float*       yrow  = out + (size_t)batch * (size_t)S;

  float4 cur = *(const float4*)xrow;                   // x[s..s+3], prefetch pipe

  for (int s0 = 0; s0 < S; s0 += 4) {
    int snext = s0 + 4; if (snext > S - 4) snext = S - 4;    // uniform clamp
    float4 nxt = *(const float4*)(xrow + snext);
    __builtin_prefetch(xrow + s0 + 64, 0, 1);                // global_prefetch_b8
    float xs[4] = {cur.x, cur.y, cur.z, cur.w};

#pragma unroll
    for (int u = 0; u < 4; ++u) {
      const float xv = xs[u];

      // B operands: h^T in bf16 hi/lo split, plane layout [n][j].
      // Lane supplies K = 16*half .. 16*half+15 -> 32 contiguous bytes.
      V16U Bhi, Blo;
      {
        const unsigned boff = (unsigned)n * 64u + (unsigned)half * 32u;
        Bhi.q[0] = *(const uint4*)(lds + LDS_HHI + boff);
        Bhi.q[1] = *(const uint4*)(lds + LDS_HHI + boff + 16);
        Blo.q[0] = *(const uint4*)(lds + LDS_HLO + boff);
        Blo.q[1] = *(const uint4*)(lds + LDS_HLO + boff + 16);
      }

      float psum = 0.0f;

#pragma unroll
      for (int g = 0; g < 2; ++g) {        // hidden-unit halves j = 16g + 8*half + r
        v8f D[4];                          // i, f, g, o gate tiles for this group
#pragma unroll
        for (int q = 0; q < 4; ++q) {
          const int mt = 2 * q + g;        // gate tiles {g, 2+g, 4+g, 6+g}
          // A operand: W_hh rows, lane = row M, K = 16h..16h+15 (contiguous)
          V16U A;
          const unsigned arow = (unsigned)(mt * 16 + n) * 64u;
          A.q[0] = *(const uint4*)(lds + LDS_A + arow + (unsigned)half * 32u);
          A.q[1] = *(const uint4*)(lds + LDS_A + arow + 16u + (unsigned)half * 32u);
          // C operand = gate bias (VGPR r holds M = r + 8*half)
          V8U Cb;
          const unsigned coff = (unsigned)(mt * 16 + 8 * half) * 4u;
          Cb.q[0] = *(const uint4*)(lds + LDS_BIAS + coff);
          Cb.q[1] = *(const uint4*)(lds + LDS_BIAS + coff + 16);

          v8f d = __builtin_amdgcn_wmma_f32_16x16x32_bf16(
              false, A.v, false, Bhi.v, (short)0, Cb.v, false, false);
          d = __builtin_amdgcn_wmma_f32_16x16x32_bf16(
              false, A.v, false, Blo.v, (short)0, d, false, false);
          D[q] = d;
        }

        // W_ih columns (C==1) for the 4 gate tiles of this group
        V8U wI, wF, wG, wO;
        {
          const unsigned o0 = (unsigned)((0 + g) * 16 + 8 * half) * 4u;
          const unsigned o1 = (unsigned)((2 + g) * 16 + 8 * half) * 4u;
          const unsigned o2 = (unsigned)((4 + g) * 16 + 8 * half) * 4u;
          const unsigned o3 = (unsigned)((6 + g) * 16 + 8 * half) * 4u;
          wI.q[0] = *(const uint4*)(lds + LDS_WIH + o0);
          wI.q[1] = *(const uint4*)(lds + LDS_WIH + o0 + 16);
          wF.q[0] = *(const uint4*)(lds + LDS_WIH + o1);
          wF.q[1] = *(const uint4*)(lds + LDS_WIH + o1 + 16);
          wG.q[0] = *(const uint4*)(lds + LDS_WIH + o2);
          wG.q[1] = *(const uint4*)(lds + LDS_WIH + o2 + 16);
          wO.q[0] = *(const uint4*)(lds + LDS_WIH + o3);
          wO.q[1] = *(const uint4*)(lds + LDS_WIH + o3 + 16);
        }

        HPK phi, plo;
#pragma unroll
        for (int r = 0; r < 8; ++r) {      // lane-local cell update, 8 units
          const float gi = D[0][r] + xv * wI.f[r];
          const float gf = D[1][r] + xv * wF.f[r];
          const float gg = D[2][r] + xv * wG.f[r];
          const float go = D[3][r] + xv * wO.f[r];
          const float cc = fsig(gf) * cst[g][r] + fsig(gi) * ftanh(gg);
          cst[g][r] = cc;
          const float h = fsig(go) * ftanh(cc);
          psum = fmaf(h, wlinv[g * 8 + r], psum);
          const __bf16 hh = (__bf16)h;     // hi/lo split keeps ~fp32 recurrence
          phi.b[r] = hh;
          plo.b[r] = (__bf16)(h - (float)hh);
        }
        // store new h for this group: j = 16g + 8*half + r -> 16 contiguous bytes
        const unsigned hoff =
            (unsigned)n * 64u + (unsigned)g * 32u + (unsigned)half * 16u;
        *(uint4*)(lds + LDS_HHI + hoff) = phi.q;
        *(uint4*)(lds + LDS_HLO + hoff) = plo.q;
      }

      // linear head: reduce the two lane-half partial dots, tanh, store
      const float tot = psum + __shfl_xor(psum, 16, 32);
      const float y   = ftanh(tot + blin);
      if (half == 0) yrow[s0 + u] = y;
    }
    cur = nxt;
  }
}

extern "C" void kernel_launch(void* const* d_in, const int* in_sizes, int n_in,
                              void* d_out, int out_size, void* d_ws, size_t ws_size,
                              hipStream_t stream) {
  const float* x     = (const float*)d_in[0];
  const float* W_ih  = (const float*)d_in[1];
  const float* W_hh  = (const float*)d_in[2];
  const float* b_ih  = (const float*)d_in[3];
  const float* b_hh  = (const float*)d_in[4];
  const float* W_lin = (const float*)d_in[5];
  const float* b_lin = (const float*)d_in[6];
  float* out = (float*)d_out;

  const int S = in_sizes[0] / 32;  // x is (B=32, C=1, S)

  lstm_wmma_kernel<<<dim3(2), dim3(32), 0, stream>>>(
      x, W_ih, W_hh, b_ih, b_hh, W_lin, b_lin, out, S);

  (void)n_in; (void)out_size; (void)d_ws; (void)ws_size;
}
